// SelfAttentionModule_65481071397415
// MI455X (gfx1250) — matrix operand order repaired and checked by reading, not compiled, not run on hardware
//
#include <hip/hip_runtime.h>
#include <hip/hip_bf16.h>
#include <math.h>

#define BB 4
#define NN 2048
#define DD 512
#define HH 8
#define HD 64
#define SCALE_F 0.044194173824159216f   // 512^-0.5 (module scales by embed_dim^-0.5)

typedef __attribute__((ext_vector_type(8)))  _Float16 v8h;
typedef __attribute__((ext_vector_type(16))) _Float16 v16h;
typedef __attribute__((ext_vector_type(8)))  float    v8f;

static __device__ __forceinline__ v8f wmma16(v16h a, v16h b, v8f c) {
  // D = A(16x32 f16) * B(32x16 f16) + C(16x16 f32)
  return __builtin_amdgcn_wmma_f32_16x16x32_f16(false, a, false, b, (short)0, c,
                                                false, false);
}

static __device__ __forceinline__ v16h pack16(v8h lo, v8h hi) {
  v16h r;
#pragma unroll
  for (int i = 0; i < 8; ++i) { r[i] = lo[i]; r[i + 8] = hi[i]; }
  return r;
}

// A-fragment (16x32 f16) from row-major [M x ld]; lane holds row m0+(lane&15),
// K-runs at k0+g*8..+7 and k0+16+g*8..+7  (ISA 16-bit A layout).
static __device__ __forceinline__ v16h load_a(const _Float16* __restrict__ base,
                                              int ld, int m0, int k0, int lane) {
  const int m = m0 + (lane & 15);
  const int g = lane >> 4;
  const _Float16* p = base + (size_t)m * ld + k0 + g * 8;
  v8h lo = *(const v8h*)p;
  v8h hi = *(const v8h*)(p + 16);
  return pack16(lo, hi);
}

// B-fragment (32x16 f16) from row-major [K x ld]; lane holds column n0+(lane&15),
// K-runs at k0+g*8..+7 and k0+16+g*8..+7 (ISA 16-bit B layout).
static __device__ __forceinline__ v16h load_b(const _Float16* __restrict__ base,
                                              int ld, int k0, int n0, int lane) {
  const int n = n0 + (lane & 15);
  const int g = lane >> 4;
  v16h r;
#pragma unroll
  for (int i = 0; i < 8; ++i) {
    r[i]     = base[(size_t)(k0 + g * 8 + i) * ld + n];
    r[i + 8] = base[(size_t)(k0 + 16 + g * 8 + i) * ld + n];
  }
  return r;
}

__global__ void f32_to_f16(const float* __restrict__ s, _Float16* __restrict__ d, int n) {
  for (int i = blockIdx.x * blockDim.x + threadIdx.x; i < n; i += gridDim.x * blockDim.x)
    d[i] = (_Float16)s[i];
}

// ---------------- QKV projection: [8192x512] x [512x1536] -----------------
// Wave = one 32x64 output tile. k-loop unrolled x2 with ping/pong fragment
// buffers: no rotation copies, WMMAs run against loads issued a stage earlier.
__global__ __launch_bounds__(256) void qkv_gemm(const _Float16* __restrict__ xh,
                                                const _Float16* __restrict__ wh,
                                                _Float16* __restrict__ q,
                                                _Float16* __restrict__ kt,
                                                _Float16* __restrict__ v) {
  const int lane = threadIdx.x & 31;
  const int wid  = blockIdx.x * (blockDim.x >> 5) + (threadIdx.x >> 5);
  const int MT   = (BB * NN) / 32;          // 256 row tiles (32 rows each)
  const int m0   = (wid % MT) * 32;
  const int n0   = (wid / MT) * 64;

  v8f acc[2][4] = {};
  v16h aA[2], bA[4], aB[2], bB[4];

  auto load_stage = [&](v16h (&a)[2], v16h (&b)[4], int k) {
    a[0] = load_a(xh, DD, m0,      k, lane);
    a[1] = load_a(xh, DD, m0 + 16, k, lane);
#pragma unroll
    for (int t = 0; t < 4; ++t) b[t] = load_b(wh, 3 * DD, k, n0 + 16 * t, lane);
  };
  auto mma_stage = [&](v16h (&a)[2], v16h (&b)[4]) {
#pragma unroll
    for (int t = 0; t < 4; ++t) {
      acc[0][t] = wmma16(a[0], b[t], acc[0][t]);
      acc[1][t] = wmma16(a[1], b[t], acc[1][t]);
    }
  };

  load_stage(aA, bA, 0);
  for (int k0 = 0; k0 < DD; k0 += 64) {
    load_stage(aB, bB, k0 + 32);
    mma_stage(aA, bA);
    load_stage(aA, bA, (k0 + 64) & (DD - 1));  // final iter wraps: harmless reload
    mma_stage(aB, bB);
  }

  const int g = lane >> 4, col = lane & 15;
  const int three = n0 >> 9;                // uniform per wave (64 | 512)
  const int bb = m0 >> 11;                  // batch index (2048 rows/batch)
#pragma unroll
  for (int half = 0; half < 2; ++half) {
#pragma unroll
    for (int t = 0; t < 4; ++t) {
      const int j  = n0 + 16 * t + col;     // column in [0,1536)
      const int h  = (j >> 6) & 7;
      const int hd = j & 63;
      const size_t bh = (size_t)bb * HH + h;
#pragma unroll
      for (int r = 0; r < 8; ++r) {
        const int n = ((m0 + 16 * half) & (NN - 1)) + r + 8 * g;
        const _Float16 val = (_Float16)acc[half][t][r];
        if (three == 0)      q [(bh * NN + n) * HD + hd] = val;   // Q  [bh,n,hd]
        else if (three == 1) kt[(bh * HD + hd) * NN + n] = val;   // K^T[bh,hd,n]
        else                 v [(bh * NN + n) * HD + hd] = val;   // V  [bh,n,hd]
      }
    }
  }
}

// ---------------- Flash attention: wave = 16 query rows ------------------
__global__ __launch_bounds__(128) void attn_fa(const _Float16* __restrict__ q,
                                               const _Float16* __restrict__ kt,
                                               const _Float16* __restrict__ v,
                                               _Float16* __restrict__ attn) {
  __shared__ _Float16 lds[4][16][32];       // per-wave P-tile staging
  const int lane = threadIdx.x & 31;
  const int w    = threadIdx.x >> 5;
  const int wid  = blockIdx.x * 4 + w;
  const int bh   = wid >> 7;                // 128 q-tiles per (b,h)
  const int q0   = (wid & 127) * 16;
  const int g    = lane >> 4, col = lane & 15;

  const _Float16* qp = q  + (size_t)bh * NN * HD;
  const _Float16* kp = kt + (size_t)bh * HD * NN;
  const _Float16* vp = v  + (size_t)bh * NN * HD;

  const v16h aq0 = load_a(qp, HD, q0, 0,  lane);   // Q rows, d=0..31
  const v16h aq1 = load_a(qp, HD, q0, 32, lane);   // Q rows, d=32..63

  float Mr[8], Lr[8];
#pragma unroll
  for (int r = 0; r < 8; ++r) { Mr[r] = -INFINITY; Lr[r] = 0.f; }
  v8f o[4] = {};

  for (int kk = 0; kk < NN; kk += 32) {
    // Issue ALL chunk loads up front: 4 K-frags and 4 V-frags, all strided 16-bit element loads.
    // V loads overlap the S WMMAs and the softmax VALU work below.
    const v16h bk00 = load_b(kp, NN, 0,  kk,      lane);
    const v16h bk01 = load_b(kp, NN, 32, kk,      lane);
    const v16h bk10 = load_b(kp, NN, 0,  kk + 16, lane);
    const v16h bk11 = load_b(kp, NN, 32, kk + 16, lane);
    v16h bv[4];
#pragma unroll
    for (int dt = 0; dt < 4; ++dt) bv[dt] = load_b(vp, HD, kk, dt * 16, lane);

    const int kkn = (kk + 32) & (NN - 1);   // prefetch next chunk
    __builtin_prefetch(kp + (size_t)lane * NN + kkn, 0, 1);
    __builtin_prefetch(vp + (size_t)(kkn + lane) * HD, 0, 1);

    v8f s0 = {}, s1 = {};
    s0 = wmma16(aq0, bk00, s0);
    s0 = wmma16(aq1, bk01, s0);
    s1 = wmma16(aq0, bk10, s1);
    s1 = wmma16(aq1, bk11, s1);

    float cm[8];
#pragma unroll
    for (int r = 0; r < 8; ++r) {
      s0[r] *= SCALE_F; s1[r] *= SCALE_F;
      cm[r] = fmaxf(s0[r], s1[r]);
    }
#pragma unroll
    for (int msk = 1; msk < 16; msk <<= 1)
#pragma unroll
      for (int r = 0; r < 8; ++r)
        cm[r] = fmaxf(cm[r], __shfl_xor(cm[r], msk, 32));   // row max (16-lane group)

    float alpha[8], cs[8];
#pragma unroll
    for (int r = 0; r < 8; ++r) {
      const float nm = fmaxf(Mr[r], cm[r]);
      alpha[r] = __expf(Mr[r] - nm);
      Mr[r] = nm;
      const float p0 = __expf(s0[r] - nm);
      const float p1 = __expf(s1[r] - nm);
      lds[w][r + 8 * g][col]      = (_Float16)p0;   // P tile -> LDS (row-major 16x32)
      lds[w][r + 8 * g][16 + col] = (_Float16)p1;
      cs[r] = p0 + p1;
    }
#pragma unroll
    for (int msk = 1; msk < 16; msk <<= 1)
#pragma unroll
      for (int r = 0; r < 8; ++r)
        cs[r] += __shfl_xor(cs[r], msk, 32);               // row sum
#pragma unroll
    for (int r = 0; r < 8; ++r) Lr[r] = Lr[r] * alpha[r] + cs[r];

    // Wave-private LDS slice; DS ops are in-order within a wave, but make the
    // store->load ordering explicit (toolchain lacks the s_wait_dscnt builtin).
    asm volatile("s_wait_dscnt 0" ::: "memory");
    const v8h plo = *(const v8h*)&lds[w][lane & 15][g * 8];
    const v8h phi = *(const v8h*)&lds[w][lane & 15][16 + g * 8];
    const v16h ap = pack16(plo, phi);                      // P as A-fragment

#pragma unroll
    for (int dt = 0; dt < 4; ++dt) {
#pragma unroll
      for (int r = 0; r < 8; ++r) o[dt][r] *= alpha[r];    // rescale running O
      o[dt] = wmma16(ap, bv[dt], o[dt]);
    }
  }

  const int bb = bh >> 3, h = bh & 7;
#pragma unroll
  for (int dt = 0; dt < 4; ++dt)
#pragma unroll
    for (int r = 0; r < 8; ++r) {
      const int n = q0 + r + 8 * g;
      attn[((size_t)(bb * NN + n)) * DD + h * HD + dt * 16 + col] =
          (_Float16)(o[dt][r] / Lr[r]);
    }
}

// ---------------- Output projection: [8192x512] x [512x512] + bias --------
__global__ __launch_bounds__(256) void out_gemm(const _Float16* __restrict__ ah,
                                                const _Float16* __restrict__ wh,
                                                const float* __restrict__ bias,
                                                float* __restrict__ out) {
  const int lane = threadIdx.x & 31;
  const int wid  = blockIdx.x * (blockDim.x >> 5) + (threadIdx.x >> 5);
  const int MT   = (BB * NN) / 32;          // 256 row tiles (32 rows each)
  const int m0   = (wid % MT) * 32;
  const int n0   = (wid / MT) * 64;

  v8f acc[2][4] = {};
  v16h aA[2], bA[4], aB[2], bB[4];

  auto load_stage = [&](v16h (&a)[2], v16h (&b)[4], int k) {
    a[0] = load_a(ah, DD, m0,      k, lane);
    a[1] = load_a(ah, DD, m0 + 16, k, lane);
#pragma unroll
    for (int t = 0; t < 4; ++t) b[t] = load_b(wh, DD, k, n0 + 16 * t, lane);
  };
  auto mma_stage = [&](v16h (&a)[2], v16h (&b)[4]) {
#pragma unroll
    for (int t = 0; t < 4; ++t) {
      acc[0][t] = wmma16(a[0], b[t], acc[0][t]);
      acc[1][t] = wmma16(a[1], b[t], acc[1][t]);
    }
  };

  load_stage(aA, bA, 0);
  for (int k0 = 0; k0 < DD; k0 += 64) {
    load_stage(aB, bB, k0 + 32);
    mma_stage(aA, bA);
    load_stage(aA, bA, (k0 + 64) & (DD - 1));
    mma_stage(aB, bB);
  }

  const int g = lane >> 4, col = lane & 15;
#pragma unroll
  for (int half = 0; half < 2; ++half)
#pragma unroll
    for (int t = 0; t < 4; ++t) {
      const int j = n0 + 16 * t + col;
      const float bj = bias[j];
#pragma unroll
      for (int r = 0; r < 8; ++r) {
        const int m = m0 + 16 * half + r + 8 * g;
        out[(size_t)m * DD + j] = acc[half][t][r] + bj;
      }
    }
}

extern "C" void kernel_launch(void* const* d_in, const int* in_sizes, int n_in,
                              void* d_out, int out_size, void* d_ws, size_t ws_size,
                              hipStream_t stream) {
  const float* x    = (const float*)d_in[0];
  const float* wqkv = (const float*)d_in[1];
  const float* wout = (const float*)d_in[2];
  const float* bout = (const float*)d_in[3];
  float* out = (float*)d_out;

  char* ws = (char*)d_ws;
  size_t off = 0;
  auto carve = [&](size_t bytes) -> char* {
    char* p = ws + off;
    off += (bytes + 255) & ~(size_t)255;
    return p;
  };
  _Float16* xh  = (_Float16*)carve((size_t)BB * NN * DD * 2);      // 8 MB
  _Float16* wqh = (_Float16*)carve((size_t)DD * 3 * DD * 2);       // 1.5 MB
  _Float16* woh = (_Float16*)carve((size_t)DD * DD * 2);           // 0.5 MB
  _Float16* q   = (_Float16*)carve((size_t)BB * HH * NN * HD * 2); // 8 MB
  _Float16* kt  = (_Float16*)carve((size_t)BB * HH * NN * HD * 2); // 8 MB
  _Float16* v   = (_Float16*)carve((size_t)BB * HH * NN * HD * 2); // 8 MB
  _Float16* at  = (_Float16*)carve((size_t)BB * NN * DD * 2);      // 8 MB

  f32_to_f16<<<2048, 256, 0, stream>>>(x,    xh,  BB * NN * DD);
  f32_to_f16<<<768,  256, 0, stream>>>(wqkv, wqh, DD * 3 * DD);
  f32_to_f16<<<256,  256, 0, stream>>>(wout, woh, DD * DD);

  // 256 row-tiles x 24 col-groups = 6144 waves, 8 waves/block
  qkv_gemm<<<768, 256, 0, stream>>>(xh, wqh, q, kt, v);
  // 32 (b,h) x 128 q-tiles = 4096 waves, 4 waves/block
  attn_fa<<<1024, 128, 0, stream>>>(q, kt, v, at);
  // 256 row-tiles x 8 col-groups = 2048 waves, 8 waves/block
  out_gemm<<<256, 256, 0, stream>>>(at, woh, bout, out);
}
